// Model_86827058856246
// MI455X (gfx1250) — matrix operand order repaired and checked by reading, not compiled, not run on hardware
//
#include <hip/hip_runtime.h>
#include <hip/hip_bf16.h>

// ---------------- CDNA5 async global->LDS availability ----------------
#if defined(__HIP_DEVICE_COMPILE__) && __has_builtin(__builtin_amdgcn_global_load_async_to_lds_b128) && __has_builtin(__builtin_amdgcn_s_wait_asynccnt)
#define USE_ASYNC_LDS 1
#else
#define USE_ASYNC_LDS 0
#endif

// ---------------- problem constants ----------------
constexpr int Bz  = 512;
constexpr int Tz  = 64;
constexpr int INz = 256;
constexpr int Ez  = 1024;
constexpr int Rz  = 8;
constexpr int HNz = 233;          // derived hidden size
constexpr int OUTz= 256;
constexpr int G4  = 4*Rz*HNz;     // 7456
constexpr int G8  = 8*HNz;        // 1864
constexpr int G4p = 7552;         // 7456 padded to 128*59
constexpr int HP  = 256;          // HN padded to 256 (K multiple of 32)
constexpr int GH  = 932;          // 4*HN, per-rule gate block

typedef __bf16 v16bf __attribute__((ext_vector_type(16)));
typedef float  v8f   __attribute__((ext_vector_type(8)));
typedef int    v4i   __attribute__((vector_size(16)));   // matches async-LDS builtin param type

__device__ __forceinline__ unsigned short f2bf(float f){
  union { float f; unsigned u; } v; v.f = f;
  unsigned r = v.u + 0x7FFFu + ((v.u >> 16) & 1u);   // round-to-nearest-even
  return (unsigned short)(r >> 16);
}

union Frag { v16bf v; uint4 q[2]; unsigned int d[8]; unsigned short u[16]; };

enum { EPI_F32 = 0, EPI_RELU_BIAS_BF16 = 1, EPI_BIAS_BF16 = 2, EPI_SCATTER_BIAS_F32 = 3 };

// ---------------- WMMA GEMM: C(MxN) = A(MxK) * B(KxN), bf16 in, f32 acc ----------------
// A is row-major bf16 (lda stride). B is pre-swizzled: B'[k/32][n][k%32] (ldb = padded N).
// M % 128 == 0, N % 128 == 0 (padded), K % 32 == 0 (padded). 256 threads = 8 waves (4x2),
// each wave computes 32x64 via 8x v_wmma_f32_16x16x32_bf16 per 32-deep K chunk.
// Staging: GLOBAL_LOAD_ASYNC_TO_LDS_B128 (ASYNCcnt) when available -> zero staging VGPRs;
// otherwise short-lived register staging placed before the compute section (no spills).
template<int EPI>
__global__ __launch_bounds__(256, 1)
void gemm_wmma(const unsigned short* __restrict__ A, long lda,
               const unsigned short* __restrict__ Bw, long ldb,
               float* __restrict__ Cf, long ldc,
               unsigned short* __restrict__ Cb, long ldcb,
               const float* __restrict__ bias,
               int M, int N, int K, int realN)
{
  constexpr int BM = 128, BN = 128, BK = 32;
  constexpr int STR = BK + 8;    // 40 ushorts = 80B rows -> every b128 access 16B aligned
  __shared__ __align__(16) unsigned short As[2][BM][STR];   // m-major, k-contiguous
  __shared__ __align__(16) unsigned short Bs[2][BN][STR];   // n-major, k-contiguous

  const int tid  = threadIdx.x;
  const int row0 = blockIdx.y * BM;
  const int col0 = blockIdx.x * BN;
  const int wid  = tid >> 5, lane = tid & 31;
  const int wm   = (wid & 3) * 32;      // wave M offset within tile
  const int wn   = (wid >> 2) * 64;     // wave N offset within tile
  const int hi   = lane >> 4, ln = lane & 15;

  v8f acc[2][4];
  #pragma unroll
  for (int i = 0; i < 2; ++i)
    #pragma unroll
    for (int j = 0; j < 4; ++j)
      #pragma unroll
      for (int q = 0; q < 8; ++q) acc[i][j][q] = 0.f;

  // global->LDS staging: 2 x 16B chunks per thread per matrix (512 chunks of 8 bf16 each)
  int ar[2], ak[2];
  #pragma unroll
  for (int c = 0; c < 2; ++c) {
    const int ch = tid + c * 256;
    ar[c] = ch >> 2;  ak[c] = (ch & 3) * 8;   // 128 rows x 4 chunks of 8
  }

  const int KT = K / BK;

  // stage K-chunk kt into LDS buffer `buf`
  auto stage = [&](int kt, int buf) {
    const int kk = kt * BK;
    #pragma unroll
    for (int c = 0; c < 2; ++c) {
      const unsigned short* ga = A  + (long)(row0 + ar[c]) * lda + kk + ak[c];
      const unsigned short* gb = Bw + ((long)kt * ldb + col0 + ar[c]) * BK + ak[c];
      unsigned short* la = &As[buf][ar[c]][ak[c]];
      unsigned short* lb = &Bs[buf][ar[c]][ak[c]];
#if USE_ASYNC_LDS
      __builtin_amdgcn_global_load_async_to_lds_b128((v4i*)ga, (v4i*)la, 0, 0);
      __builtin_amdgcn_global_load_async_to_lds_b128((v4i*)gb, (v4i*)lb, 0, 0);
#else
      const uint4 va = *(const uint4*)ga;
      const uint4 vb = *(const uint4*)gb;
      *(uint4*)la = va;
      *(uint4*)lb = vb;
#endif
    }
  };
  auto stage_wait = [&]() {
#if USE_ASYNC_LDS
    __builtin_amdgcn_s_wait_asynccnt(0);
#endif
    __syncthreads();
  };

  stage(0, 0);
  stage_wait();

  for (int kt = 0; kt < KT; ++kt) {
    const int  cur = kt & 1;
    const bool nxt = (kt + 1) < KT;
    if (nxt) stage(kt + 1, 1 - cur);   // DMA (or short-lived regs) into the back buffer

    // A fragments: lane = row M=ln (hi selects K half); VGPRs 0-3 = K 0..7 (+8*hi),
    // VGPRs 4-7 = K 16..23 (+8*hi)  ->  2x ds_load_b128 per fragment
    Frag a[2];
    #pragma unroll
    for (int i = 0; i < 2; ++i) {
      const int mr = wm + i * 16 + ln;
      a[i].q[0] = *(const uint4*)(&As[cur][mr][hi * 8]);
      a[i].q[1] = *(const uint4*)(&As[cur][mr][16 + hi * 8]);
    }
    // B fragments: lane = col N=ln (hi selects K half); VGPRs 0-3 = K 0..7 (+8*hi),
    // VGPRs 4-7 = K 16..23 (+8*hi) -> 2x ds_load_b128
    Frag bf[4];
    #pragma unroll
    for (int j = 0; j < 4; ++j) {
      const int nc = wn + j * 16 + ln;
      bf[j].q[0] = *(const uint4*)(&Bs[cur][nc][hi * 8]);
      bf[j].q[1] = *(const uint4*)(&Bs[cur][nc][16 + hi * 8]);
    }
    #pragma unroll
    for (int i = 0; i < 2; ++i)
      #pragma unroll
      for (int j = 0; j < 4; ++j)
        acc[i][j] = __builtin_amdgcn_wmma_f32_16x16x32_bf16(
            false, a[i].v, false, bf[j].v, (short)0, acc[i][j], false, false);

    if (nxt) stage_wait();
  }

  // epilogue: C/D 16x16 layout -> element q: row = q + 8*hi, col = ln
  #pragma unroll
  for (int i = 0; i < 2; ++i) {
    #pragma unroll
    for (int j = 0; j < 4; ++j) {
      const int n = col0 + wn + j * 16 + ln;
      #pragma unroll
      for (int q = 0; q < 8; ++q) {
        const int m = row0 + wm + i * 16 + q + hi * 8;
        const float v = acc[i][j][q];
        if constexpr (EPI == EPI_F32) {
          Cf[(long)m * ldc + n] = v;
        } else if constexpr (EPI == EPI_RELU_BIAS_BF16) {
          const float t = (n < realN) ? fmaxf(v + bias[n], 0.f) : 0.f;
          Cb[(long)m * ldcb + n] = f2bf(t);
        } else if constexpr (EPI == EPI_BIAS_BF16) {
          const float t = (n < realN) ? (v + bias[n]) : 0.f;
          Cb[(long)m * ldcb + n] = f2bf(t);
        } else { // scatter (T,B,OUT) row m = t*B+b  ->  out[b][t][n]
          const float t = v + bias[n];
          const int b_ = m & (Bz - 1);
          const int t_ = m >> 9;
          Cf[((long)b_ * Tz + t_) * OUTz + n] = t;
        }
      }
    }
  }
}

// ---------------- fused per-timestep kernel (score + softmax + LSTM + rule mix) ----------------
__global__ __launch_bounds__(256)
void step_kernel(const float* __restrict__ Ci, const float* __restrict__ Ch,
                 const float* __restrict__ sw, const float* __restrict__ c_prev,
                 float* __restrict__ c_new, unsigned short* __restrict__ h_bf,
                 unsigned short* __restrict__ hs_bf, float* __restrict__ scores_out,
                 int t)
{
  const int b    = blockIdx.x;
  const int tid  = threadIdx.x;
  const int r    = tid >> 5;          // 8 waves == 8 rules
  const int lane = tid & 31;
  __shared__ float sS[Rz];

  const float* cir = Ci + (long)b * G4p;
  const float* chr = Ch + (long)b * G4p;

  // score s[b][r] = sum_k relu(feat) * scorer_w[r][k]
  float acc = 0.f;
  const float* swr = sw + r * G8;
  const int gb0 = r * GH;
  for (int k = lane; k < G8; k += 32) {
    float v = (k < GH) ? cir[gb0 + k] : chr[gb0 + (k - GH)];
    v = fmaxf(v, 0.f);
    acc += v * swr[k];
  }
  #pragma unroll
  for (int off = 16; off > 0; off >>= 1) acc += __shfl_down(acc, off, 32);
  if (lane == 0) sS[r] = acc;
  __syncthreads();

  // softmax over rules (registers, every thread)
  float sv[Rz]; float mx = -1e30f;
  #pragma unroll
  for (int rr = 0; rr < Rz; ++rr) { sv[rr] = sS[rr]; mx = fmaxf(mx, sv[rr]); }
  float sum = 0.f;
  #pragma unroll
  for (int rr = 0; rr < Rz; ++rr) { sv[rr] = expf(sv[rr] - mx); sum += sv[rr]; }
  const float inv = 1.f / sum;
  #pragma unroll
  for (int rr = 0; rr < Rz; ++rr) sv[rr] *= inv;
  if (tid < Rz) scores_out[((long)b * Tz + t) * Rz + tid] = sv[tid];

  // LSTM gates + rule-weighted reduction, one hidden unit per thread
  const int h = tid;
  float hsum = 0.f, csum = 0.f;
  if (h < HNz) {
    const float cp = c_prev[(long)b * HP + h];
    #pragma unroll
    for (int rr = 0; rr < Rz; ++rr) {
      const int gb = rr * GH;
      const float pi = cir[gb +           h] + chr[gb +           h];
      const float pf = cir[gb +   HNz +   h] + chr[gb +   HNz +   h];
      const float po = cir[gb + 2*HNz +   h] + chr[gb + 2*HNz +   h];
      const float pg = cir[gb + 3*HNz +   h] + chr[gb + 3*HNz +   h];
      const float ig = 1.f / (1.f + expf(-pi));
      const float fg = 1.f / (1.f + expf(-pf));
      const float og = 1.f / (1.f + expf(-po));
      const float gg = tanhf(pg);
      const float ct = cp * fg + ig * gg;
      const float ht = og * tanhf(ct);
      hsum += sv[rr] * ht;
      csum += sv[rr] * ct;
    }
  }
  c_new[(long)b * HP + h] = csum;
  const unsigned short hv = f2bf(hsum);        // 0 in the pad region
  h_bf[(long)b * HP + h] = hv;
  hs_bf[((long)t * Bz + b) * HP + h] = hv;
}

// ---------------- pack f32 -> bf16, row-major with zero padding (A-side) ----------------
__global__ void pack_pad(const float* __restrict__ src, unsigned short* __restrict__ dst,
                         int K, int N, int Np, long total)
{
  const long i = (long)blockIdx.x * 256 + threadIdx.x;
  if (i >= total) return;
  const int k = (int)(i / Np), n = (int)(i % Np);
  dst[i] = (k < K && n < N) ? f2bf(src[(long)k * N + n]) : (unsigned short)0;
}

// ---------------- pack f32 -> bf16, B-swizzled: dst[k/32][n][k%32] ----------------
__global__ void pack_bswz(const float* __restrict__ src, unsigned short* __restrict__ dst,
                          int K, int N, int Np, long total)
{
  const long i = (long)blockIdx.x * 256 + threadIdx.x;
  if (i >= total) return;
  const long nk32 = (long)Np * 32;
  const int  kc  = (int)(i / nk32);
  const long rem = i % nk32;
  const int  n   = (int)(rem >> 5);
  const int  ko  = (int)(rem & 31);
  const int  k   = kc * 32 + ko;
  dst[i] = (k < K && n < N) ? f2bf(src[(long)k * N + n]) : (unsigned short)0;
}

__global__ void zero_f32(float* p, long n)          { long i = (long)blockIdx.x*256 + threadIdx.x; if (i < n) p[i] = 0.f; }
__global__ void zero_u16(unsigned short* p, long n) { long i = (long)blockIdx.x*256 + threadIdx.x; if (i < n) p[i] = 0;  }

// ---------------- launch ----------------
extern "C" void kernel_launch(void* const* d_in, const int* in_sizes, int n_in,
                              void* d_out, int out_size, void* d_ws, size_t ws_size,
                              hipStream_t stream)
{
  (void)in_sizes; (void)n_in; (void)out_size; (void)ws_size;
  const float* x      = (const float*)d_in[0];
  const float* enc_w1 = (const float*)d_in[2];
  const float* enc_b1 = (const float*)d_in[3];
  const float* enc_w2 = (const float*)d_in[4];
  const float* enc_b2 = (const float*)d_in[5];
  const float* i2h_w  = (const float*)d_in[6];
  const float* h2h_w  = (const float*)d_in[7];
  const float* scorer = (const float*)d_in[8];
  const float* dec_w1 = (const float*)d_in[9];
  const float* dec_b1 = (const float*)d_in[10];
  const float* dec_w2 = (const float*)d_in[11];
  const float* dec_b2 = (const float*)d_in[12];

  float* outp    = (float*)d_out;
  float* scoresp = outp + (long)Bz * Tz * OUTz;

  const long BT = (long)Bz * Tz;                  // 32768
  char* p = (char*)d_ws;
  auto carve = [&](size_t bytes) -> char* {
    char* r = p; p += (bytes + 255) & ~(size_t)255; return r;
  };
  unsigned short* x_bf  = (unsigned short*)carve(BT * INz * 2);
  unsigned short* w1b   = (unsigned short*)carve((size_t)INz * Ez  * 2);
  unsigned short* w2b   = (unsigned short*)carve((size_t)Ez  * Ez  * 2);
  unsigned short* i2hb  = (unsigned short*)carve((size_t)Ez  * G4p * 2);
  unsigned short* h2hb  = (unsigned short*)carve((size_t)HP  * G4p * 2);
  unsigned short* dw1b  = (unsigned short*)carve((size_t)HP  * HP  * 2);
  unsigned short* dw2b  = (unsigned short*)carve((size_t)HP  * HP  * 2);
  unsigned short* h1b   = (unsigned short*)carve(BT * Ez * 2);
  unsigned short* encb  = (unsigned short*)carve(BT * Ez * 2);
  float*          Ci    = (float*)carve((size_t)Bz * G4p * 4);
  float*          Ch    = (float*)carve((size_t)Bz * G4p * 4);
  float*          c_a   = (float*)carve((size_t)Bz * HP * 4);
  float*          c_b   = (float*)carve((size_t)Bz * HP * 4);
  unsigned short* h_bf  = (unsigned short*)carve((size_t)Bz * HP * 2);
  unsigned short* hs_bf = (unsigned short*)carve(BT * HP * 2);
  unsigned short* d1b   = (unsigned short*)carve(BT * HP * 2);

  auto packA = [&](const float* s, unsigned short* d, int K, int N, int Kp, int Np) {
    const long total = (long)Kp * Np;
    pack_pad<<<dim3((unsigned)((total + 255) / 256)), dim3(256), 0, stream>>>(s, d, K, N, Np, total);
  };
  auto packB = [&](const float* s, unsigned short* d, int K, int N, int Kp, int Np) {
    const long total = (long)Kp * Np;
    pack_bswz<<<dim3((unsigned)((total + 255) / 256)), dim3(256), 0, stream>>>(s, d, K, N, Np, total);
  };

  // input packing (A-side, row-major bf16)
  packA(x, x_bf, (int)BT, INz, (int)BT, INz);
  // weight packing (B-side, K-chunk swizzled bf16, zero-padded)
  packB(enc_w1, w1b,  INz, Ez,  INz, Ez);
  packB(enc_w2, w2b,  Ez,  Ez,  Ez,  Ez);
  packB(i2h_w,  i2hb, Ez,  G4,  Ez,  G4p);
  packB(h2h_w,  h2hb, HNz, G4,  HP,  G4p);
  packB(dec_w1, dw1b, HNz, HNz, HP,  HP);
  packB(dec_w2, dw2b, HNz, OUTz,HP,  HP);

  // state init
  zero_u16<<<dim3((Bz * HP + 255) / 256), dim3(256), 0, stream>>>(h_bf, (long)Bz * HP);
  zero_f32<<<dim3((Bz * HP + 255) / 256), dim3(256), 0, stream>>>(c_a, (long)Bz * HP);
  zero_f32<<<dim3((Bz * HP + 255) / 256), dim3(256), 0, stream>>>(c_b, (long)Bz * HP);

  const dim3 blk(256);
  // encoder
  gemm_wmma<EPI_RELU_BIAS_BF16><<<dim3(Ez / 128, (unsigned)(BT / 128)), blk, 0, stream>>>(
      x_bf, INz, w1b, Ez, nullptr, 0, h1b, Ez, enc_b1, (int)BT, Ez, INz, Ez);
  gemm_wmma<EPI_BIAS_BF16><<<dim3(Ez / 128, (unsigned)(BT / 128)), blk, 0, stream>>>(
      h1b, Ez, w2b, Ez, nullptr, 0, encb, Ez, enc_b2, (int)BT, Ez, Ez, Ez);

  // recurrence
  float* cp = c_a;
  float* cn = c_b;
  for (int t = 0; t < Tz; ++t) {
    // i2h: rows of enc for step t are strided (row b lives at b*T + t)
    gemm_wmma<EPI_F32><<<dim3(G4p / 128, Bz / 128), blk, 0, stream>>>(
        encb + (long)t * Ez, (long)Tz * Ez, i2hb, G4p, Ci, G4p, nullptr, 0, nullptr,
        Bz, G4p, Ez, G4p);
    // h2h
    gemm_wmma<EPI_F32><<<dim3(G4p / 128, Bz / 128), blk, 0, stream>>>(
        h_bf, HP, h2hb, G4p, Ch, G4p, nullptr, 0, nullptr,
        Bz, G4p, HP, G4p);
    // fused gating / scoring step
    step_kernel<<<dim3(Bz), blk, 0, stream>>>(Ci, Ch, scorer, cp, cn, h_bf, hs_bf, scoresp, t);
    float* tmp = cp; cp = cn; cn = tmp;
  }

  // decoder
  gemm_wmma<EPI_RELU_BIAS_BF16><<<dim3(HP / 128, (unsigned)(BT / 128)), blk, 0, stream>>>(
      hs_bf, HP, dw1b, HP, nullptr, 0, d1b, HP, dec_b1, (int)BT, HP, HP, HNz);
  gemm_wmma<EPI_SCATTER_BIAS_F32><<<dim3(HP / 128, (unsigned)(BT / 128)), blk, 0, stream>>>(
      d1b, HP, dw2b, HP, outp, 0, nullptr, 0, dec_b2, (int)BT, HP, HP, OUTz);
}
